// QuantMasoaNet_86930138071384
// MI455X (gfx1250) — matrix operand order repaired and checked by reading, not compiled, not run on hardware
//
#include <hip/hip_runtime.h>
#include <hip/hip_bf16.h>
#include <math.h>

// ---------------------------------------------------------------------------
// MI455X (gfx1250) fused MLP stack: 6x [GEMM1 -> GELU -> GEMM2 -> +res -> LN]
// Compute-bound (AI ~2000 FLOP/B) => bf16 WMMA (16x16x32) with a 3-term hi/lo
// split (bf16x3) for near-fp32 accuracy at ~8/3x the f32-WMMA rate. Weights
// pre-split once into workspace. x tile staged to LDS via the Tensor Data
// Mover (TENSORcnt) where available; direct loads otherwise.
// ---------------------------------------------------------------------------

#define DIMK   2048
#define HIDK   8192
#define TOKK   8192      // 4 * 2048 tokens
#define NBLK   6
#define MT     16        // tokens per workgroup
#define NWAVE  16        // waves per workgroup (wave32 -> 512 threads)
#define TPB    (NWAVE * 32)
#define NC     (NWAVE * 16)   // 256 hidden columns per chunk

typedef __attribute__((ext_vector_type(16))) __bf16       v16bf;
typedef __attribute__((ext_vector_type(8)))  float        v8f;
typedef __attribute__((ext_vector_type(8)))  unsigned int v8u;
typedef __attribute__((ext_vector_type(4)))  unsigned int u32x4;
typedef __attribute__((ext_vector_type(8)))  int          i32x8;
typedef __attribute__((ext_vector_type(4)))  int          i32x4;

#if defined(__gfx1250__) && __has_builtin(__builtin_amdgcn_tensor_load_to_lds) && \
    __has_builtin(__builtin_amdgcn_s_wait_tensorcnt)
#define USE_TDM 1
#else
#define USE_TDM 0
#endif

static __device__ __forceinline__ v16bf mkbf(uint4 a, uint4 b) {
  v8u t;
  t[0] = a.x; t[1] = a.y; t[2] = a.z; t[3] = a.w;
  t[4] = b.x; t[5] = b.y; t[6] = b.z; t[7] = b.w;
  return __builtin_bit_cast(v16bf, t);
}

// (neg_a, A, neg_b, B, c_mod, C, reuse_a, reuse_b); RA/RB as immediates.
template <bool RA, bool RB>
static __device__ __forceinline__ v8f wmma_bf16(v16bf a, v16bf b, v8f c) {
  return __builtin_amdgcn_wmma_f32_16x16x32_bf16(false, a, false, b, (short)0, c,
                                                 RA, RB);
}

// bf16x3 triple with operand-reuse chaining: Ah*Bl -> Ah*Bh(RA) -> Al*Bh(RB)
static __device__ __forceinline__ v8f wmma3(v16bf Ah, v16bf Al, v16bf Bh,
                                            v16bf Bl, v8f c) {
  c = wmma_bf16<false, false>(Ah, Bl, c);
  c = wmma_bf16<true,  false>(Ah, Bh, c);
  c = wmma_bf16<false, true >(Al, Bh, c);
  return c;
}

static __device__ __forceinline__ void split_f32(float v, unsigned short& h,
                                                 unsigned short& l) {
  unsigned u = __float_as_uint(v);
  h = (unsigned short)(u >> 16);                       // truncated bf16 prefix
  float hf = __uint_as_float((unsigned)h << 16);
  float lo = v - hf;                                   // exact residual
  l = (unsigned short)(__float_as_uint(lo) >> 16);
}

// ---------------------------------------------------------------------------
// Prep: split f32 weights into hi/lo bf16 planes (one ~70us HBM streaming
// pass; negligible vs the ~ms of WMMA work it enables).
// ---------------------------------------------------------------------------
__global__ void split_weights_kernel(const float* __restrict__ w,
                                     unsigned short* __restrict__ hi,
                                     unsigned short* __restrict__ lo,
                                     size_t n) {
  size_t i = ((size_t)blockIdx.x * blockDim.x + threadIdx.x) * 4;
  if (i + 4 <= n) {
    float4 v = *(const float4*)(w + i);
    unsigned short h0, l0, h1, l1, h2, l2, h3, l3;
    split_f32(v.x, h0, l0); split_f32(v.y, h1, l1);
    split_f32(v.z, h2, l2); split_f32(v.w, h3, l3);
    ushort4 hv; hv.x = h0; hv.y = h1; hv.z = h2; hv.w = h3;
    ushort4 lv; lv.x = l0; lv.y = l1; lv.z = l2; lv.w = l3;
    *(ushort4*)(hi + i) = hv;
    *(ushort4*)(lo + i) = lv;
  }
}

// ---------------------------------------------------------------------------
// Fused block kernel. Grid = TOKK/MT workgroups; 16 waves each.
// Dynamic LDS: xs_hi/xs_lo (16x2048 bf16 each, aliases the f32 TDM landing
// zone), hs_hi/hs_lo (16x256 bf16), 32-float LN scratch => 147584 B
// (2 WGs per 320KB WGP).
// ---------------------------------------------------------------------------
__global__ __launch_bounds__(TPB, 1)
void mlp_block_kernel(const float* __restrict__ xin, float* __restrict__ xout,
                      const unsigned short* __restrict__ w1hi,
                      const unsigned short* __restrict__ w1lo,
                      const float* __restrict__ b1,
                      const unsigned short* __restrict__ w2hi,
                      const unsigned short* __restrict__ w2lo,
                      const float* __restrict__ b2,
                      const float* __restrict__ lng,
                      const float* __restrict__ lnb, int has_ln) {
  extern __shared__ char smem[];
  unsigned short* xs_hi = (unsigned short*)smem;           // [16][2048]
  unsigned short* xs_lo = xs_hi + MT * DIMK;               // [16][2048]
  unsigned short* hs_hi = xs_lo + MT * DIMK;               // [16][256]
  unsigned short* hs_lo = hs_hi + MT * NC;                 // [16][256]
  float*          red   = (float*)(hs_lo + MT * NC);       // [32]

  const int tid    = threadIdx.x;
  const int wave   = tid >> 5;
  const int lane   = tid & 31;
  const int laneN  = lane & 15;
  const int laneHi = lane >> 4;       // 0: lanes 0-15, 1: lanes 16-31
  const int m0     = blockIdx.x * MT;

  // ---- stage x tile as hi/lo bf16 into LDS -------------------------------
#if USE_TDM
  {
    // TDM: one 2D descriptor pulls the 16x2048 f32 tile into LDS bytes
    // [0, 128K), which exactly aliases the xs_hi+xs_lo bf16 planes.
    if (wave == 0) {
      const unsigned long long ga =
          (unsigned long long)(uintptr_t)(xin + (size_t)m0 * DIMK);
      u32x4 g0;
      g0[0] = 1u;                                   // count=1, user mode
      g0[1] = 0u;                                   // lds_addr = base of LDS
      g0[2] = (unsigned)(ga & 0xffffffffu);         // global_addr[31:0]
      g0[3] = (unsigned)((ga >> 32) & 0x01ffffffu)  // global_addr[56:32]
            | (2u << 30);                           // type = 2 ("image")
      i32x8 g1;
      g1[0] = (int)(2u << 16);       // data_size=2 (4B); no pad/iter/multicast
      g1[1] = (int)(2048u << 16);    // tensor_dim0[15:0] = 2048
      g1[2] = (int)(8192u << 16);    // tensor_dim0[31:16]=0, tensor_dim1 lo=8192
      g1[3] = (int)(2048u << 16);    // tensor_dim1 hi=0, tile_dim0 = 2048
      g1[4] = 16;                    // tile_dim1 = 16 rows, tile_dim2 = 0
      g1[5] = 2048;                  // tensor_dim0_stride[31:0]
      g1[6] = 0;
      g1[7] = 0;
      i32x4 z4 = (i32x4)0;
#if __clang_major__ >= 23
      i32x8 z8 = (i32x8)0;
      __builtin_amdgcn_tensor_load_to_lds(g0, g1, z4, z4, z8, 0);
#else
      __builtin_amdgcn_tensor_load_to_lds(g0, g1, z4, z4, 0);
#endif
      __builtin_amdgcn_s_wait_tensorcnt(0);
    }
    __syncthreads();
    // In-place f32 -> (hi,lo) bf16 plane conversion, register-buffered.
    const float* xf = (const float*)smem;
    float tmp[64];
#pragma unroll
    for (int j = 0; j < 64; ++j) tmp[j] = xf[tid + j * TPB];
    __syncthreads();
#pragma unroll
    for (int j = 0; j < 64; ++j) {
      unsigned short h, l;
      split_f32(tmp[j], h, l);
      xs_hi[tid + j * TPB] = h;
      xs_lo[tid + j * TPB] = l;
    }
  }
#else
  for (int i = tid * 4; i < MT * DIMK; i += TPB * 4) {
    float4 v = *(const float4*)(xin + (size_t)m0 * DIMK + i);
    unsigned short h0, l0, h1, l1, h2, l2, h3, l3;
    split_f32(v.x, h0, l0); split_f32(v.y, h1, l1);
    split_f32(v.z, h2, l2); split_f32(v.w, h3, l3);
    ushort4 hv; hv.x = h0; hv.y = h1; hv.z = h2; hv.w = h3;
    ushort4 lv; lv.x = l0; lv.y = l1; lv.z = l2; lv.w = l3;
    *(ushort4*)(xs_hi + i) = hv;
    *(ushort4*)(xs_lo + i) = lv;
  }
#endif
  __syncthreads();

  // ---- f32 accumulators: 8 C-tiles (this wave owns y cols wave*128..+127) -
  v8f acc[8];
#pragma unroll
  for (int t = 0; t < 8; ++t) acc[t] = (v8f)0.0f;

  // ---- hidden-chunk loop --------------------------------------------------
  for (int c = 0; c < HIDK / NC; ++c) {
    const int hbase = c * NC + wave * 16;   // this wave's h tile columns

    // GEMM1: h_tile(16x16) = x(16x2048) * W1[hbase..+16)^T, bf16x3 split
    v8f hacc = (v8f)0.0f;
    const unsigned short* w1r_h =
        w1hi + (size_t)(hbase + laneN) * DIMK + laneHi * 16;
    const unsigned short* w1r_l =
        w1lo + (size_t)(hbase + laneN) * DIMK + laneHi * 16;
#pragma unroll 2
    for (int kk = 0; kk < DIMK; kk += 32) {
      const int ka = kk + laneHi * 16;
      // A operand (contiguous K order: VGPR0-7 = K[ka..ka+15])
      const unsigned short* ap = xs_hi + laneN * DIMK + ka;
      v16bf Ah = mkbf(*(const uint4*)ap, *(const uint4*)(ap + 8));
      const unsigned short* alp = xs_lo + laneN * DIMK + ka;
      v16bf Al = mkbf(*(const uint4*)alp, *(const uint4*)(alp + 8));
      // B operand (same contiguous K order as A: lanes0-15 K=0-15, lanes16-31 K=16-31)
      v16bf Bh = mkbf(*(const uint4*)(w1r_h + kk), *(const uint4*)(w1r_h + kk + 8));
      v16bf Bl = mkbf(*(const uint4*)(w1r_l + kk), *(const uint4*)(w1r_l + kk + 8));
      __builtin_prefetch(w1r_h + kk + 64, 0, 3);
      hacc = wmma3(Ah, Al, Bh, Bl, hacc);
    }

    // bias + exact-erf GELU, split to hi/lo bf16, publish to LDS
    const float bias1 = b1[hbase + laneN];
#pragma unroll
    for (int r = 0; r < 8; ++r) {
      float hv = hacc[r] + bias1;
      float g = 0.5f * hv * (1.0f + erff(hv * 0.70710678118654752f));
      unsigned short h, l;
      split_f32(g, h, l);
      const int row = r + laneHi * 8;
      hs_hi[row * NC + wave * 16 + laneN] = h;
      hs_lo[row * NC + wave * 16 + laneN] = l;
    }
    __syncthreads();

    // GEMM2 accumulate: y[:, wave*128..+128) += gelu_chunk * W2_chunk^T
    const int kglob0 = c * NC;
#pragma unroll 1
    for (int kk2 = 0; kk2 < NC; kk2 += 32) {
      const int ka2 = kk2 + laneHi * 16;
      const unsigned short* ap = hs_hi + laneN * NC + ka2;
      v16bf Ah = mkbf(*(const uint4*)ap, *(const uint4*)(ap + 8));
      const unsigned short* alp = hs_lo + laneN * NC + ka2;
      v16bf Al = mkbf(*(const uint4*)alp, *(const uint4*)(alp + 8));
      const int kglob = kglob0 + kk2 + laneHi * 16;
#pragma unroll
      for (int t = 0; t < 8; ++t) {
        const int col = wave * 128 + t * 16 + laneN;
        const unsigned short* w2r_h = w2hi + (size_t)col * HIDK + kglob;
        const unsigned short* w2r_l = w2lo + (size_t)col * HIDK + kglob;
        v16bf Bh = mkbf(*(const uint4*)w2r_h, *(const uint4*)(w2r_h + 8));
        v16bf Bl = mkbf(*(const uint4*)w2r_l, *(const uint4*)(w2r_l + 8));
        acc[t] = wmma3(Ah, Al, Bh, Bl, acc[t]);
      }
    }
    __syncthreads();   // protect hs before next chunk overwrites it
  }

  // ---- bias2 + residual ---------------------------------------------------
#pragma unroll
  for (int t = 0; t < 8; ++t) {
    const int col = wave * 128 + t * 16 + laneN;
    const float bias2 = b2[col];
#pragma unroll
    for (int r = 0; r < 8; ++r) {
      const int row = m0 + r + laneHi * 8;
      acc[t][r] += bias2 + xin[(size_t)row * DIMK + col];
    }
  }

  // ---- LayerNorm (blocks 0..4): LDS ds_add_f32 row reductions -------------
  if (has_ln) {
    if (tid < 32) red[tid] = 0.0f;
    __syncthreads();
#pragma unroll
    for (int r = 0; r < 8; ++r) {
      float s = 0.0f, q = 0.0f;
#pragma unroll
      for (int t = 0; t < 8; ++t) {
        float v = acc[t][r];
        s += v;
        q += v * v;
      }
      const int lm = r + laneHi * 8;
      atomicAdd(&red[lm], s);
      atomicAdd(&red[16 + lm], q);
    }
    __syncthreads();
#pragma unroll
    for (int r = 0; r < 8; ++r) {
      const int lm = r + laneHi * 8;
      const float mu  = red[lm] * (1.0f / DIMK);
      const float msq = red[16 + lm] * (1.0f / DIMK);
      const float rs  = rsqrtf(msq - mu * mu + 1e-5f);
#pragma unroll
      for (int t = 0; t < 8; ++t) {
        const int col = wave * 128 + t * 16 + laneN;
        acc[t][r] = (acc[t][r] - mu) * rs * lng[col] + lnb[col];
      }
    }
  }

  // ---- store --------------------------------------------------------------
#pragma unroll
  for (int t = 0; t < 8; ++t) {
    const int col = wave * 128 + t * 16 + laneN;
#pragma unroll
    for (int r = 0; r < 8; ++r) {
      const int row = m0 + r + laneHi * 8;
      xout[(size_t)row * DIMK + col] = acc[t][r];
    }
  }
}

// ---------------------------------------------------------------------------
// Host launcher. Workspace layout (assumes ws_size >= ~832 MB):
//   [0]      W1 hi bf16 (6*8192*2048 ush) | [+nW] W1 lo | [+2nW] W2 hi |
//   [+3nW]   W2 lo | [+4nW] x ping buffer (8192*2048 f32)
// Activation chain: in -> ws -> out -> ws -> out -> ws -> out (block 5 = out).
// ---------------------------------------------------------------------------
extern "C" void kernel_launch(void* const* d_in, const int* in_sizes, int n_in,
                              void* d_out, int out_size, void* d_ws,
                              size_t ws_size, hipStream_t stream) {
  const float* x   = (const float*)d_in[0];
  const float* W1  = (const float*)d_in[1];
  const float* B1  = (const float*)d_in[2];
  const float* W2  = (const float*)d_in[3];
  const float* B2  = (const float*)d_in[4];
  const float* LNG = (const float*)d_in[5];
  const float* LNB = (const float*)d_in[6];
  float* out = (float*)d_out;

  const size_t nW = (size_t)NBLK * HIDK * DIMK;
  unsigned short* w1hi = (unsigned short*)d_ws;
  unsigned short* w1lo = w1hi + nW;
  unsigned short* w2hi = w1lo + nW;
  unsigned short* w2lo = w2hi + nW;
  float* xping = (float*)(w2lo + nW);

  {
    const int thr = 256;
    const size_t nq = nW / 4;
    const int blk = (int)((nq + thr - 1) / thr);
    split_weights_kernel<<<blk, thr, 0, stream>>>(W1, w1hi, w1lo, nW);
    split_weights_kernel<<<blk, thr, 0, stream>>>(W2, w2hi, w2lo, nW);
  }

  const size_t shmem = (size_t)(2 * MT * DIMK + 2 * MT * NC) * sizeof(unsigned short)
                     + 32 * sizeof(float);   // 147584 B dynamic LDS

  const float* cur = x;
  for (int b = 0; b < NBLK; ++b) {
    float* o = (b & 1) ? out : xping;        // odd blocks -> d_out; b5 -> d_out
    const int ln = (b < NBLK - 1) ? 1 : 0;
    mlp_block_kernel<<<TOKK / MT, TPB, shmem, stream>>>(
        cur, o,
        w1hi + (size_t)b * HIDK * DIMK, w1lo + (size_t)b * HIDK * DIMK,
        B1 + (size_t)b * HIDK,
        w2hi + (size_t)b * (size_t)DIMK * HIDK,
        w2lo + (size_t)b * (size_t)DIMK * HIDK,
        B2 + (size_t)b * DIMK,
        ln ? LNG + (size_t)b * DIMK : LNG,
        ln ? LNB + (size_t)b * DIMK : LNB, ln);
    cur = o;
  }
}
